// TestModel_80582176408255
// MI455X (gfx1250) — 8-bit matrix operand order repaired and checked by reading, compiled, not run on hardware
//
#include <hip/hip_runtime.h>
#include <hip/hip_bf16.h>
#include <stdint.h>

#ifndef __has_builtin
#define __has_builtin(x) 0
#endif

#if __has_builtin(__builtin_amdgcn_global_load_async_to_lds_b64) && \
    __has_builtin(__builtin_amdgcn_global_load_async_to_lds_b128)
#define HAVE_ASYNC_LDS 1
#else
#define HAVE_ASYNC_LDS 0
#endif

typedef int vi2_t __attribute__((vector_size(8)));
typedef int vi4_t __attribute__((vector_size(16)));

__device__ __forceinline__ void copy_g2l_b64(const unsigned char* g, unsigned char* l) {
#if HAVE_ASYNC_LDS
    __builtin_amdgcn_global_load_async_to_lds_b64(
        (__attribute__((address_space(1))) vi2_t*)g,
        (__attribute__((address_space(3))) vi2_t*)l, 0, 0);
#else
    *(uint64_t*)l = *(const uint64_t*)g;
#endif
}
__device__ __forceinline__ void copy_g2l_b128(const unsigned char* g, unsigned char* l) {
#if HAVE_ASYNC_LDS
    __builtin_amdgcn_global_load_async_to_lds_b128(
        (__attribute__((address_space(1))) vi4_t*)g,
        (__attribute__((address_space(3))) vi4_t*)l, 0, 0);
#else
    *(int4*)l = *(const int4*)g;
#endif
}
__device__ __forceinline__ void wait_async_copies() {
#if HAVE_ASYNC_LDS
#if __has_builtin(__builtin_amdgcn_s_wait_asynccnt)
    __builtin_amdgcn_s_wait_asynccnt(0);
#else
    asm volatile("s_wait_asynccnt 0x0" ::: "memory");
#endif
#endif
}

__device__ __forceinline__ unsigned char f32_to_e4m3(float x) {
    unsigned sign = (__float_as_uint(x) >> 31) << 7;
    float a = fabsf(x);
    a = fminf(a, 448.0f);
    if (a < 0.015625f) {
        int q = (int)rintf(a * 512.0f);
        if (q >= 8) return (unsigned char)(sign | 0x08u);
        return (unsigned char)(sign | (unsigned)q);
    }
    unsigned bits = __float_as_uint(a);
    int e = (int)((bits >> 23) & 0xffu) - 127;
    unsigned m = bits & 0x7fffffu;
    unsigned mant = m >> 20;
    unsigned rest = m & 0xfffffu;
    if (rest > 0x80000u || (rest == 0x80000u && (mant & 1u))) {
        mant++;
        if (mant == 8u) { mant = 0u; e++; }
    }
    if (e > 8) return (unsigned char)(sign | 0x7Eu);
    return (unsigned char)(sign | ((unsigned)(e + 7) << 3) | mant);
}

__global__ __launch_bounds__(256)
void row_norm_quant_kernel(const float* __restrict__ src,
                           float* __restrict__ resid_out,
                           const float* __restrict__ w,
                           unsigned char* __restrict__ qx,
                           float* __restrict__ srow,
                           float* __restrict__ yout,
                           int do_relu) {
    const int H = 4096;
    const int row = blockIdx.x;
    const int tid = threadIdx.x;
    __shared__ float red[256];

    float v[16];
    float ss = 0.0f;
#pragma unroll
    for (int i = 0; i < 16; ++i) {
        float t = src[(size_t)row * H + tid + i * 256];
        if (do_relu) t = fmaxf(t, 0.0f);
        v[i] = t;
        ss += t * t;
    }
    if (resid_out) {
#pragma unroll
        for (int i = 0; i < 16; ++i)
            resid_out[(size_t)row * H + tid + i * 256] = v[i];
    }

    red[tid] = ss;
    __syncthreads();
#pragma unroll
    for (int s = 128; s > 0; s >>= 1) {
        if (tid < s) red[tid] += red[tid + s];
        __syncthreads();
    }
    const float rstd = rsqrtf(red[0] * (1.0f / 4096.0f) + 1e-6f);
    __syncthreads();

    float y[16];
    float amax = 0.0f;
#pragma unroll
    for (int i = 0; i < 16; ++i) {
        y[i] = v[i] * rstd * w[tid + i * 256];
        amax = fmaxf(amax, fabsf(y[i]));
    }

    if (yout) {
#pragma unroll
        for (int i = 0; i < 16; ++i)
            yout[(size_t)row * H + tid + i * 256] = y[i];
    }

    if (qx) {
        red[tid] = amax;
        __syncthreads();
#pragma unroll
        for (int s = 128; s > 0; s >>= 1) {
            if (tid < s) red[tid] = fmaxf(red[tid], red[tid + s]);
            __syncthreads();
        }
        const float sc = fmaxf(red[0] * (1.0f / 448.0f), 1e-12f);
        if (tid == 0) srow[row] = sc;
        const float inv = 1.0f / sc;
#pragma unroll
        for (int i = 0; i < 16; ++i) {
            float q = y[i] * inv;
            q = fminf(fmaxf(q, -448.0f), 448.0f);
            qx[(size_t)row * H + tid + i * 256] = f32_to_e4m3(q);
        }
    }
}

__global__ void amax_zero_kernel(unsigned* amax) { amax[0] = 0u; }

__global__ __launch_bounds__(256)
void weight_amax_kernel(const float* __restrict__ w, unsigned* __restrict__ amax, int n4) {
    __shared__ float red[256];
    const int tid = threadIdx.x;
    float m = 0.0f;
    const float4* w4 = (const float4*)w;
    for (int i = blockIdx.x * 256 + tid; i < n4; i += gridDim.x * 256) {
        float4 v = w4[i];
        m = fmaxf(m, fmaxf(fmaxf(fabsf(v.x), fabsf(v.y)), fmaxf(fabsf(v.z), fabsf(v.w))));
    }
    red[tid] = m;
    __syncthreads();
#pragma unroll
    for (int s = 128; s > 0; s >>= 1) {
        if (tid < s) red[tid] = fmaxf(red[tid], red[tid + s]);
        __syncthreads();
    }
    if (tid == 0) atomicMax(amax, __float_as_uint(red[0]));
}

__global__ __launch_bounds__(256)
void weight_quant_kernel(const float* __restrict__ w, unsigned char* __restrict__ qw,
                         const unsigned* __restrict__ amax, int n4) {
    const float ws = fmaxf(__uint_as_float(amax[0]) * (1.0f / 448.0f), 1e-12f);
    const float inv = 1.0f / ws;
    const int i = blockIdx.x * 256 + threadIdx.x;
    if (i >= n4) return;
    float4 v = ((const float4*)w)[i];
    unsigned r = (unsigned)f32_to_e4m3(fminf(fmaxf(v.x * inv, -448.0f), 448.0f))
               | ((unsigned)f32_to_e4m3(fminf(fmaxf(v.y * inv, -448.0f), 448.0f)) << 8)
               | ((unsigned)f32_to_e4m3(fminf(fmaxf(v.z * inv, -448.0f), 448.0f)) << 16)
               | ((unsigned)f32_to_e4m3(fminf(fmaxf(v.w * inv, -448.0f), 448.0f)) << 24);
    ((unsigned*)qw)[i] = r;
}

typedef __attribute__((ext_vector_type(16))) int   v16i;
typedef __attribute__((ext_vector_type(8)))  float v8f;

#define LSTRIDE     80
#define TILE_BYTES  (32 * LSTRIDE)
#define SLAB_BYTES  (8 * TILE_BYTES)
#define STAGE_BYTES (2 * SLAB_BYTES)
#define SMEM_BYTES  (2 * STAGE_BYTES)

__device__ __forceinline__ void stage_tiles(const unsigned char* __restrict__ qx,
                                            const unsigned char* __restrict__ qw,
                                            int Mblk, int Nblk, int kk,
                                            unsigned char* sA, unsigned char* sB,
                                            int tid) {
    const int K = 4096;
#pragma unroll
    for (int i = 0; i < 8; ++i) {
        int id = i * 256 + tid;
        int m = id >> 4, cidx = id & 15;
        int b = cidx >> 3, c = (cidx >> 1) & 3, kh = cidx & 1;
        const unsigned char* g = qx + (size_t)(Mblk + m) * K + kk + b * 64 + (c >> 1) * 32 + kh * 16 + (c & 1) * 8;
        unsigned char* l = sA + (m >> 4) * TILE_BYTES
                              + (kh * 16 + (m & 15)) * LSTRIDE + b * 32 + c * 8;
        copy_g2l_b64(g, l);
    }
#pragma unroll
    for (int i = 0; i < 4; ++i) {
        int id = i * 256 + tid;
        int n = id >> 3, cidx = id & 7;
        int cc = cidx >> 1, kh = cidx & 1;
        const unsigned char* g = qw + (size_t)(Nblk + n) * K + kk + cc * 32 + kh * 16;
        unsigned char* l = sB + (n >> 4) * TILE_BYTES
                              + (kh * 16 + (n & 15)) * LSTRIDE + cc * 16;
        copy_g2l_b128(g, l);
    }
}

__device__ __forceinline__ v16i load_frag(const unsigned char* l) {
    v16i f;
#pragma unroll
    for (int q = 0; q < 4; ++q) {
        int4 v = *(const int4*)(l + q * 16);
        f[q * 4 + 0] = v.x;
        f[q * 4 + 1] = v.y;
        f[q * 4 + 2] = v.z;
        f[q * 4 + 3] = v.w;
    }
    return f;
}

__global__ __launch_bounds__(256)
void fp8_gemm_add_kernel(const unsigned char* __restrict__ qx,
                         const unsigned char* __restrict__ qw,
                         const float* __restrict__ srow,
                         const unsigned* __restrict__ amax,
                         float* __restrict__ resid) {
    extern __shared__ unsigned char smem[];
    const int K = 4096;
    const int tid   = threadIdx.x;
    const int lane  = tid & 31;
    const int wave  = tid >> 5;
    const int wrow  = wave >> 1;
    const int wcol  = wave & 1;
    const int Mblk  = blockIdx.y * 128;
    const int Nblk  = blockIdx.x * 128;
    const int Mbase = Mblk + wrow * 32;
    const int Nbase = Nblk + wcol * 64;
    const int lrow  = lane & 15;
    const int khalf = lane >> 4;

    v8f acc[2][4];
#pragma unroll
    for (int mt = 0; mt < 2; ++mt)
#pragma unroll
        for (int nt = 0; nt < 4; ++nt)
#pragma unroll
            for (int j = 0; j < 8; ++j) acc[mt][nt][j] = 0.0f;

    stage_tiles(qx, qw, Mblk, Nblk, 0, smem, smem + SLAB_BYTES, tid);
    wait_async_copies();
    __syncthreads();

    for (int kk = 0; kk < K; kk += 128) {
        const int cur = (kk >> 7) & 1;
        unsigned char* sA = smem + cur * STAGE_BYTES;
        unsigned char* sB = sA + SLAB_BYTES;

        if (kk + 128 < K) {
            unsigned char* nA = smem + (cur ^ 1) * STAGE_BYTES;
            stage_tiles(qx, qw, Mblk, Nblk, kk + 128, nA, nA + SLAB_BYTES, tid);
        }

        v16i afrag[2];
#pragma unroll
        for (int mt = 0; mt < 2; ++mt)
            afrag[mt] = load_frag(sA + (wrow * 2 + mt) * TILE_BYTES + lane * LSTRIDE);

#pragma unroll
        for (int nt = 0; nt < 4; ++nt) {
            v16i bfrag = load_frag(sB + (wcol * 4 + nt) * TILE_BYTES + lane * LSTRIDE);
            acc[0][nt] = __builtin_amdgcn_wmma_f32_16x16x128_fp8_fp8(
                afrag[0], bfrag, (short)0, acc[0][nt], false, false);
            acc[1][nt] = __builtin_amdgcn_wmma_f32_16x16x128_fp8_fp8(
                afrag[1], bfrag, (short)0, acc[1][nt], false, false);
        }

        wait_async_copies();
        __syncthreads();
    }

    const float ws = fmaxf(__uint_as_float(amax[0]) * (1.0f / 448.0f), 1e-12f);
#pragma unroll
    for (int mt = 0; mt < 2; ++mt) {
        float sm[8];
#pragma unroll
        for (int j = 0; j < 8; ++j)
            sm[j] = srow[Mbase + mt * 16 + khalf * 8 + j] * ws;
#pragma unroll
        for (int nt = 0; nt < 4; ++nt) {
            const int col = Nbase + nt * 16 + lrow;
#pragma unroll
            for (int j = 0; j < 8; ++j) {
                const int row = Mbase + mt * 16 + khalf * 8 + j;
                resid[(size_t)row * K + col] += acc[mt][nt][j] * sm[j];
            }
        }
    }
}

extern "C" void kernel_launch(void* const* d_in, const int* in_sizes, int n_in,
                              void* d_out, int out_size, void* d_ws, size_t ws_size,
                              hipStream_t stream) {
    const int H = 4096, T = 4096;
    const float* x = (const float*)d_in[0];
    const float* nw[4] = { (const float*)d_in[1], (const float*)d_in[2],
                           (const float*)d_in[3], (const float*)d_in[4] };
    const float* W[3]  = { (const float*)d_in[5], (const float*)d_in[6],
                           (const float*)d_in[7] };

    char* base = (char*)d_ws;
    float*         resid = (float*)base;         base += (size_t)T * H * 4;
    unsigned char* qx    = (unsigned char*)base; base += (size_t)T * H;
    unsigned char* qw    = (unsigned char*)base; base += (size_t)H * H;
    float*         srow  = (float*)base;         base += (size_t)T * 4;
    unsigned*      amax  = (unsigned*)base;

    const int n4 = H * H / 4;

    row_norm_quant_kernel<<<T, 256, 0, stream>>>(x, resid, nw[0], qx, srow,
                                                 nullptr,  1);

    for (int l = 0; l < 3; ++l) {
        amax_zero_kernel<<<1, 1, 0, stream>>>(amax);
        weight_amax_kernel<<<2048, 256, 0, stream>>>(W[l], amax, n4);
        weight_quant_kernel<<<(n4 + 255) / 256, 256, 0, stream>>>(W[l], qw, amax, n4);
        fp8_gemm_add_kernel<<<dim3(H / 128, T / 128), 256, SMEM_BYTES, stream>>>(
            qx, qw, srow, amax, resid);
        if (l < 2) {
            row_norm_quant_kernel<<<T, 256, 0, stream>>>(
                resid, nullptr, nw[l + 1], qx, srow, nullptr, 0);
        } else {
            row_norm_quant_kernel<<<T, 256, 0, stream>>>(
                resid, nullptr, nw[3], nullptr, nullptr, (float*)d_out, 0);
        }
    }
}
